// EquivariantProductBasisBlock_36464272343563
// MI455X (gfx1250) — matrix operand order repaired and checked by reading, not compiled, not run on hardware
//
#include <hip/hip_runtime.h>

typedef __bf16 bf16;
typedef __bf16 bf16x8 __attribute__((ext_vector_type(8)));
typedef __bf16 v16bf  __attribute__((ext_vector_type(16)));
typedef float  v8f    __attribute__((ext_vector_type(8)));

#define INV_SQRT_C 0.08838834764831845f
#define NNODES 4096

// ---------------- workspace layout (bytes) ----------------
// A1  : bf16 [432 x 96]   folded U3_1|U2_1; row = (m*9+a)*16 + b (b<9 real, rest 0)
// A0  : bf16 [144 x 64]   folded U3_0|U2_0; row = a*16 + b
// T0  : bf16 [128 x 128]  Wlin_0 transposed: T0[d][c]
// T1  : bf16 [128 x 128]  Wlin_1 transposed
// Bws : bf16 4 planes of [4096 x 128]  (plane0 = L0; planes 1..3 = L1 m=0..2)
#define WS_A1  0
#define WS_A0  82944
#define WS_T0  101376
#define WS_T1  134144
#define WS_B   166912

__device__ inline v16bf load_frag2x8(const bf16* p1, const bf16* p2) {
  bf16x8 a = *(const bf16x8*)p1;
  bf16x8 b = *(const bf16x8*)p2;
  v16bf r;
#pragma unroll
  for (int j = 0; j < 8; ++j) { r[j] = a[j]; r[j + 8] = b[j]; }
  return r;
}

__device__ inline v8f wmma_bf16(v16bf a, v16bf b, v8f c) {
  return __builtin_amdgcn_wmma_f32_16x16x32_bf16(false, a, false, b, (short)0, c,
                                                 false, false);
}

// xor-16 lane swap for wave32: ds_swizzle group-of-32, xor_mask=0x10, and_mask=0x1f
__device__ inline float xor16_swap(float v) {
  return __int_as_float(__builtin_amdgcn_ds_swizzle(__float_as_int(v), 0x401F));
}

// ---------------- kernel 1: fold U tensors + transpose Wlin ----------------
__global__ __launch_bounds__(256) void mace_prep_kernel(
    const float* __restrict__ U3_1, const float* __restrict__ U2_1,
    const float* __restrict__ U3_0, const float* __restrict__ U2_0,
    const float* __restrict__ Wl0,  const float* __restrict__ Wl1,
    bf16* __restrict__ A1, bf16* __restrict__ A0,
    bf16* __restrict__ T0, bf16* __restrict__ T1) {
  const int n1 = 432 * 96, n0 = 144 * 64, nt = 128 * 128;
  int idx = blockIdx.x * 256 + threadIdx.x;
  if (idx < n1) {
    int row = idx / 96, k = idx - row * 96;
    int r2 = row >> 4, b = row & 15;      // r2 = m*9+a
    float v = 0.f;
    if (b < 9) {
      int orig = r2 * 9 + b;              // (m,a,b) flat row
      if (k < 63)      { int i = k / 7, p = k - i * 7; v = U3_1[(orig * 9 + i) * 7 + p]; }
      else if (k < 66) { v = U2_1[orig * 3 + (k - 63)]; }
    }
    A1[idx] = (bf16)v;
  } else if (idx < n1 + n0) {
    int j = idx - n1;
    int row = j / 64, k = j - row * 64;
    int r2 = row >> 4, b = row & 15;      // r2 = a
    float v = 0.f;
    if (b < 9) {
      int orig = r2 * 9 + b;
      if (k < 45)      { int i = k / 5, p = k - i * 5; v = U3_0[(orig * 9 + i) * 5 + p]; }
      else if (k < 47) { v = U2_0[orig * 2 + (k - 45)]; }
    }
    A0[j] = (bf16)v;
  } else if (idx < n1 + n0 + nt) {
    int j = idx - n1 - n0;
    int d = j >> 7, c = j & 127;
    T0[j] = (bf16)Wl0[c * 128 + d];
  } else if (idx < n1 + n0 + 2 * nt) {
    int j = idx - n1 - n0 - nt;
    int d = j >> 7, c = j & 127;
    T1[j] = (bf16)Wl1[c * 128 + d];
  }
}

// ---------------- GEMM group: NS strips, ct-outer, fused b-contraction ----
// Tile row tile rt == r2 (each (m,a) padded to 16 rows); lane half 0 holds
// b=0..7, half 1 holds b=8 (b>=9 rows are zero in A => acc exactly 0 there,
// so the xv values multiplied against them may be any finite garbage).
template <int KT, int KPAD, int NS>
__device__ inline void gemm_group(const bf16* __restrict__ Ag,
                                  const bf16* __restrict__ XW,
                                  float* __restrict__ o2,
                                  const float* __restrict__ xsh,
                                  const int* rts, int lo, int half) {
  v16bf afrag[NS][KT];
#pragma unroll
  for (int s = 0; s < NS; ++s)
#pragma unroll
    for (int kt = 0; kt < KT; ++kt) {
      const bf16* p1 = Ag + (rts[s] * 16 + lo) * KPAD + kt * 32 + 8 * half;
      afrag[s][kt] = load_frag2x8(p1, p1 + 16);
    }
#pragma unroll
  for (int ct = 0; ct < 8; ++ct) {
    const int col = ct * 16 + lo;
    v16bf bfrag[KT];
#pragma unroll
    for (int kt = 0; kt < KT; ++kt) {
      const bf16* bp = XW + col * KPAD + kt * 32 + 8 * half;
      bfrag[kt] = load_frag2x8(bp, bp + 16);
    }
    // Branch-free x loads: half0 -> x[0..3],x[4..7]; half1 -> x[8..11] (pads 0)
    // plus x[4..7] which only meets zero accumulators.
    const float4 xa = *(const float4*)(xsh + col * 12 + 8 * half);
    const float4 xb = *(const float4*)(xsh + col * 12 + 4);
    float xv[8];
    xv[0] = xa.x; xv[1] = xa.y; xv[2] = xa.z; xv[3] = xa.w;
    xv[4] = xb.x; xv[5] = xb.y; xv[6] = xb.z; xv[7] = xb.w;
#pragma unroll
    for (int s = 0; s < NS; ++s) {
      v8f acc = {};
#pragma unroll
      for (int kt = 0; kt < KT; ++kt)
        acc = wmma_bf16(afrag[s][kt], bfrag[kt], acc);
      float sum = 0.f;
#pragma unroll
      for (int v = 0; v < 8; ++v) sum += acc[v] * xv[v];
      sum += xor16_swap(sum);             // combine b=0..7 with b=8
      o2[rts[s] * 128 + col] = sum;       // both halves write identical value
    }
  }
}

// ---------------- kernel 2: per-node symmetric contraction ----------------
__global__ __launch_bounds__(256) void mace_node_kernel(
    const float* __restrict__ xg, const float* __restrict__ yg,
    const float* __restrict__ W3_0, const float* __restrict__ W2_0,
    const float* __restrict__ W1_0, const float* __restrict__ U1_0,
    const float* __restrict__ W3_1, const float* __restrict__ W2_1,
    const float* __restrict__ W1_1, const float* __restrict__ U1_1,
    const bf16* __restrict__ A1, const bf16* __restrict__ A0,
    bf16* __restrict__ Bws) {
  __shared__ __align__(16) float xsh[128 * 12];     // x[c][i], stride 12, pads 0
  __shared__ __align__(16) bf16  XW1s[128 * 96];    // B operand L1, c-major [c][k]
  __shared__ __align__(16) bf16  XW0s[128 * 64];    // B operand L0
  __shared__ __align__(16) float o2_1[27 * 128];    // out2 L1 [(m,a)][c]
  __shared__ __align__(16) float o2_0[9 * 128];     // out2 L0 [a][c]

  const int n = blockIdx.x;
  const int tid = threadIdx.x;

  // element id (uniform across the block -> scalarizes)
  int e = 0;
#pragma unroll
  for (int j = 0; j < 10; ++j)
    if (yg[n * 10 + j] > 0.5f) e = j;

  for (int idx = tid; idx < 128 * 12; idx += 256) {
    int c = idx / 12, i = idx - c * 12;
    xsh[idx] = (i < 9) ? xg[n * 1152 + c * 9 + i] : 0.f;
  }
  __syncthreads();

  // Build node-dependent B operands: XW[c][k] = x[c,i]*w3[p,c] (k=i*P3+p), then w2 rows.
  for (int idx = tid; idx < 128 * 96; idx += 256) {
    int c = idx / 96, k = idx - c * 96;
    float v = 0.f;
    if (k < 63)      { int i = k / 7, p = k - i * 7; v = xsh[c * 12 + i] * W3_1[(e * 7 + p) * 128 + c]; }
    else if (k < 66) { v = W2_1[(e * 3 + (k - 63)) * 128 + c]; }
    XW1s[idx] = (bf16)v;
  }
  for (int idx = tid; idx < 128 * 64; idx += 256) {
    int c = idx / 64, k = idx - c * 64;
    float v = 0.f;
    if (k < 45)      { int i = k / 5, p = k - i * 5; v = xsh[c * 12 + i] * W3_0[(e * 5 + p) * 128 + c]; }
    else if (k < 47) { v = W2_0[(e * 2 + (k - 45)) * 128 + c]; }
    XW0s[idx] = (bf16)v;
  }
  __syncthreads();

  const int w = tid >> 5, l = tid & 31;
  const int lo = l & 15, half = l >> 4;

  // L1: 27 strips. Waves 0..2 take 4, waves 3..7 take 3.
  if (w < 3) {
    const int rts[4] = {w, w + 8, w + 16, w + 24};
    gemm_group<3, 96, 4>(A1, XW1s, o2_1, xsh, rts, lo, half);
  } else {
    const int rts[3] = {w, w + 8, w + 16};
    gemm_group<3, 96, 3>(A1, XW1s, o2_1, xsh, rts, lo, half);
  }
  // L0: 9 strips. Wave 0 takes {0,8}, waves 1..7 take {w}.
  if (w == 0) {
    const int rts[2] = {0, 8};
    gemm_group<2, 64, 2>(A0, XW0s, o2_0, xsh, rts, lo, half);
  } else {
    const int rts[1] = {w};
    gemm_group<2, 64, 1>(A0, XW0s, o2_0, xsh, rts, lo, half);
  }
  __syncthreads();

  // Final a-contraction (+ U1*w1 term) -> B[n,c,m] planes (bf16).
  for (int idx = tid; idx < 512; idx += 256) {
    int plane = idx >> 7, c = idx & 127;
    float sacc = 0.f;
    if (plane == 0) {
      float w1c = W1_0[e * 128 + c];
#pragma unroll
      for (int a = 0; a < 9; ++a)
        sacc += (o2_0[a * 128 + c] + U1_0[a] * w1c) * xsh[c * 12 + a];
    } else {
      int m = plane - 1;
      float w1c = W1_1[e * 128 + c];
#pragma unroll
      for (int a = 0; a < 9; ++a)
        sacc += (o2_1[(m * 9 + a) * 128 + c] + U1_1[m * 9 + a] * w1c) * xsh[c * 12 + a];
    }
    Bws[(size_t)plane * (NNODES * 128) + n * 128 + c] = (bf16)sacc;
  }
}

// ---------------- kernel 3: per-irrep Linear (WMMA) + sc residual ----------------
__global__ __launch_bounds__(256) void mace_linear_kernel(
    const bf16* __restrict__ Bws, const bf16* __restrict__ T0,
    const bf16* __restrict__ T1, const float* __restrict__ sc,
    float* __restrict__ out) {
  const int rt = blockIdx.x;          // node row-tile (16 nodes)
  const int plane = blockIdx.y;       // 0=L0, 1..3=L1 m
  const int w = threadIdx.x >> 5, l = threadIdx.x & 31;
  const int lo = l & 15, half = l >> 4;
  const int ct = w;                   // 8 waves = 8 col-tiles of d

  const bf16* Ap = Bws + (size_t)plane * (NNODES * 128);
  const bf16* Wt = (plane == 0) ? T0 : T1;
  const int rowA = rt * 16 + lo;

  v8f acc = {};
#pragma unroll
  for (int kt = 0; kt < 4; ++kt) {
    const bf16* ap = Ap + rowA * 128 + kt * 32 + 8 * half;
    v16bf afrag = load_frag2x8(ap, ap + 16);
    const bf16* bp = Wt + (ct * 16 + lo) * 128 + kt * 32 + 8 * half;
    v16bf bfrag = load_frag2x8(bp, bp + 16);
    acc = wmma_bf16(afrag, bfrag, acc);
  }
  const int d = ct * 16 + lo;
  const int ocol = (plane == 0) ? d : (128 + d * 3 + (plane - 1));
#pragma unroll
  for (int v = 0; v < 8; ++v) {
    int n = rt * 16 + v + 8 * half;
    out[(size_t)n * 512 + ocol] = acc[v] * INV_SQRT_C + sc[(size_t)n * 512 + ocol];
  }
}

extern "C" void kernel_launch(void* const* d_in, const int* in_sizes, int n_in,
                              void* d_out, int out_size, void* d_ws, size_t ws_size,
                              hipStream_t stream) {
  const float* node_feats = (const float*)d_in[0];
  const float* sc         = (const float*)d_in[1];
  const float* node_attrs = (const float*)d_in[2];
  const float* U3_0 = (const float*)d_in[3];
  const float* U2_0 = (const float*)d_in[4];
  const float* U1_0 = (const float*)d_in[5];
  const float* W3_0 = (const float*)d_in[6];
  const float* W2_0 = (const float*)d_in[7];
  const float* W1_0 = (const float*)d_in[8];
  const float* Wl0  = (const float*)d_in[9];
  const float* U3_1 = (const float*)d_in[10];
  const float* U2_1 = (const float*)d_in[11];
  const float* U1_1 = (const float*)d_in[12];
  const float* W3_1 = (const float*)d_in[13];
  const float* W2_1 = (const float*)d_in[14];
  const float* W1_1 = (const float*)d_in[15];
  const float* Wl1  = (const float*)d_in[16];

  char* ws = (char*)d_ws;
  bf16* A1  = (bf16*)(ws + WS_A1);
  bf16* A0  = (bf16*)(ws + WS_A0);
  bf16* T0  = (bf16*)(ws + WS_T0);
  bf16* T1  = (bf16*)(ws + WS_T1);
  bf16* Bws = (bf16*)(ws + WS_B);
  float* out = (float*)d_out;

  mace_prep_kernel<<<326, 256, 0, stream>>>(U3_1, U2_1, U3_0, U2_0, Wl0, Wl1,
                                            A1, A0, T0, T1);
  mace_node_kernel<<<NNODES, 256, 0, stream>>>(node_feats, node_attrs,
                                               W3_0, W2_0, W1_0, U1_0,
                                               W3_1, W2_1, W1_1, U1_1,
                                               A1, A0, Bws);
  mace_linear_kernel<<<dim3(256, 4), 256, 0, stream>>>(Bws, T0, T1, sc, out);
}
